// FilteredNoise_4638564680519
// MI455X (gfx1250) — matrix operand order repaired and checked by reading, not compiled, not run on hardware
//
#include <hip/hip_runtime.h>

// ---------------------------------------------------------------------------
// FilteredNoise synthesis for MI455X (gfx1250, wave32, WMMA).
//
//   Step 1 (WMMA GEMM): ir[r,t] = sum_k X[r,k] * C[t,k]
//       X = noise_filter  [128000 x 65]  (f32 -> f16 in LDS)
//       C = windowed cosine basis (constant), stored transposed [144 x 96] f16
//       -> v_wmma_f32_16x16x32_f16, 3 k-steps per tile, f32 accumulation
//   Step 2 (VALU conv + overlap-add as a gather): each output sample
//       p = 64*f0 + j0 sums partial convolutions of frames f0, f0-1, f0-2
//       (exactly 129 MACs per sample, no atomics, coalesced stores).
// ---------------------------------------------------------------------------

typedef __attribute__((ext_vector_type(16))) _Float16 v16h;
typedef __attribute__((ext_vector_type(8)))  _Float16 v8h;
typedef __attribute__((ext_vector_type(8)))  float    v8f;

#define B_      32
#define F_      4000
#define L_      65
#define HOP     64
#define N_IR    129                     // 2L-1
#define KLEN    192                     // hop + 2L-2
#define ROWS    (B_ * F_)               // 128000
#define KPAD    96                      // K: 65 -> 96 (3 WMMA k-steps of 32)
#define NPAD    144                     // t: 129 -> 144 (9 n-tiles of 16)
#define IRS     132                     // ir row stride in floats (pad 129->132)
#define ASTR    104                     // LDS A-tile stride in halves (bank-friendly)
#define OUTLEN  ((F_ - 1) * HOP + KLEN) // 256128
#define GROUPS  (OUTLEN / HOP)          // 4002
#define GAIN    0.01f

// -------------------------- kernel 1: basis ---------------------------------
// basisT[t*KPAD + k] = win[t] * (1/129) * (k==0 ? 1 : 2) * cos(2*pi*k*(t-64)/129)
// Time-major / K-contiguous so a WMMA B-fragment is one contiguous 32B load.
__global__ void fn_basis_kernel(_Float16* __restrict__ basisT) {
    int idx = blockIdx.x * blockDim.x + threadIdx.x;
    if (idx >= NPAD * KPAD) return;
    int t = idx / KPAD;
    int k = idx % KPAD;
    float v = 0.0f;
    if (t < N_IR && k < L_) {
        const float PI2 = 6.28318530717958647692f;
        float win   = 0.5f * (1.0f - cosf(PI2 * (float)t / 129.0f));
        float c     = cosf(PI2 * (float)k * (float)(t - 64) / 129.0f);
        float scale = (k == 0) ? (1.0f / 129.0f) : (2.0f / 129.0f);
        v = win * scale * c;
    }
    basisT[idx] = (_Float16)v;
}

// -------------------------- kernel 2: WMMA GEMM -----------------------------
// One block = 9 waves = one 16-row M-tile; wave w owns N-tile w (columns 16w..16w+15).
__global__ __launch_bounds__(288) void fn_ir_gemm_kernel(
    const float* __restrict__ X,         // [ROWS x 65]
    const _Float16* __restrict__ basisT, // [NPAD x KPAD]
    float* __restrict__ ir)              // [ROWS x IRS]
{
    __shared__ _Float16 Ash[16 * ASTR];

    const int mtile = blockIdx.x;        // 0..7999
    const int tid   = threadIdx.x;

    // Cooperative load of the 16 x 96 A-tile (f32 -> f16), zero-padded K.
    for (int i = tid; i < 16 * KPAD; i += 288) {
        int r = i / KPAD, k = i % KPAD;
        float v = (k < L_) ? X[(mtile * 16 + r) * L_ + k] : 0.0f;
        Ash[r * ASTR + k] = (_Float16)v;
    }
    __syncthreads();

    const int wave = tid >> 5;           // 0..8 == n_tile
    const int lane = tid & 31;
    const int m    = lane & 15;
    const int hi   = lane >> 4;          // half-wave select
    const int ncol = wave * 16 + (lane & 15);

    v8f acc = {};
    #pragma unroll
    for (int kk = 0; kk < KPAD; kk += 32) {
        // A fragment, contiguous K order (matches B):
        //   lanes 0-15 : K kk..kk+15
        //   lanes 16-31: K kk+16..kk+31
        v8h alo = *(const v8h*)&Ash[m * ASTR + kk + hi * 16];
        v8h ahi = *(const v8h*)&Ash[m * ASTR + kk + hi * 16 + 8];
        v16h a;
        #pragma unroll
        for (int i = 0; i < 8; i++) { a[i] = alo[i]; a[i + 8] = ahi[i]; }

        // B fragment: VGPR i holds K rows (hoff+2i, hoff+2i+1) at column ncol,
        // hoff = 0 (lanes 0-15) / 16 (lanes 16-31) -> one contiguous 32B load.
        v16h b = *(const v16h*)&basisT[ncol * KPAD + kk + hi * 16];

        acc = __builtin_amdgcn_wmma_f32_16x16x32_f16(
            /*neg_a=*/false, a, /*neg_b=*/false, b,
            /*c_mod=*/(short)0, acc, /*reuse_a=*/false, /*reuse_b=*/false);
    }

    // D layout: lane group hi selects rows v / v+8; column = ncol.
    if (ncol < N_IR) {
        const int rbase = mtile * 16 + hi * 8;
        #pragma unroll
        for (int v = 0; v < 8; v++)
            ir[(size_t)(rbase + v) * IRS + ncol] = acc[v];
    }
}

// ---------------- kernel 3: per-row convolution + overlap-add gather --------
// Block = 64 threads = one 64-sample output group (b, f0). Output sample
// j0 gathers from frames f0 (j=j0), f0-1 (j=j0+64), f0-2 (j=j0+128):
// 129 MACs total, all operands in LDS, no atomics.
__global__ __launch_bounds__(64) void fn_conv_ola_kernel(
    const float* __restrict__ noise,     // [B_ x F_ x HOP]
    const float* __restrict__ ir,        // [ROWS x IRS]
    float* __restrict__ out)             // [B_ x OUTLEN]
{
    __shared__ float nsh[3][HOP];
    __shared__ float irsh[3][IRS];

    const int f0 = blockIdx.x;           // 0..4001
    const int b  = blockIdx.y;
    const int t  = threadIdx.x;          // 0..63

    #pragma unroll
    for (int d = 0; d < 3; d++) {
        int  f  = f0 - d;
        bool ok = (f >= 0) && (f < F_);
        size_t row = (size_t)(b * F_ + (ok ? f : 0));
        nsh[d][t] = ok ? (noise[row * HOP + t] * 2.0f - 1.0f) : 0.0f;
        for (int c = t; c < N_IR; c += HOP)
            irsh[d][c] = ok ? ir[row * IRS + c] : 0.0f;
    }
    __syncthreads();

    float acc = 0.0f;
    for (int s = 0; s <= t; s++)        acc += nsh[0][s] * irsh[0][t - s];       // frame f0
    for (int s = 0; s < HOP; s++)       acc += nsh[1][s] * irsh[1][t + 64 - s];  // frame f0-1
    for (int s = t; s < HOP; s++)       acc += nsh[2][s] * irsh[2][t + 128 - s]; // frame f0-2

    out[(size_t)b * OUTLEN + (size_t)f0 * HOP + t] = acc * GAIN;
}

// ---------------------------------------------------------------------------
extern "C" void kernel_launch(void* const* d_in, const int* in_sizes, int n_in,
                              void* d_out, int out_size, void* d_ws, size_t ws_size,
                              hipStream_t stream) {
    const float* X     = (const float*)d_in[0];   // noise_filter [32,4000,65]
    const float* noise = (const float*)d_in[1];   // noise        [32,4000,64]
    float*       out   = (float*)d_out;           // [32, 256128]

    // Workspace layout: basisT (f16, 27,648 B) | ir (f32, 128000*132*4 B)
    _Float16* basisT = (_Float16*)d_ws;
    float*    ir     = (float*)((char*)d_ws + 32768);

    fn_basis_kernel<<<(NPAD * KPAD + 255) / 256, 256, 0, stream>>>(basisT);
    fn_ir_gemm_kernel<<<ROWS / 16, 288, 0, stream>>>(X, basisT, ir);
    fn_conv_ola_kernel<<<dim3(GROUPS, B_), 64, 0, stream>>>(noise, ir, out);
}
